// BT4_61529701483326
// MI455X (gfx1250) — compile-verified, matrix operand order checked by reading, not run on hardware
//
#include <hip/hip_runtime.h>
#include <cmath>

// ---------------------------------------------------------------------------
// BT4 forward pass for gfx1250 (MI455X). All GEMMs via v_wmma_f32_16x16x32_bf16
// (wave32 WMMA, fp32 accumulate). Operands staged in padded bf16 scratch so the
// WMMA inner loops are branch-free, use 128-bit loads, and are software-
// pipelined (next k-step fragments loaded before current WMMAs issue).
// ---------------------------------------------------------------------------

typedef __attribute__((ext_vector_type(16))) __bf16 v16bf;
typedef __attribute__((ext_vector_type(8)))  float  v8f;

struct U4   { unsigned int x, y, z, w; };
struct Frag { U4 lo, hi; };

__device__ __forceinline__ v16bf load_frag(const unsigned short* p) {
  // p is 16B aligned; elements [0..7] at p, [8..15] at p+16 (ISA A/B layout:
  // lanes hold K = kbase+{0..7} and kbase+{16..23}).
  Frag f;
  f.lo = *(const U4*)(p);
  f.hi = *(const U4*)(p + 16);
  return __builtin_bit_cast(v16bf, f);
}

__device__ __forceinline__ v8f zero8() {
  v8f v;
  #pragma unroll
  for (int i = 0; i < 8; i++) v[i] = 0.f;
  return v;
}

__device__ __forceinline__ unsigned short f2bf(float f) {
  unsigned int u = __builtin_bit_cast(unsigned int, f);
  unsigned int r = u + 0x7FFFu + ((u >> 16) & 1u);   // round-to-nearest-even
  return (unsigned short)(r >> 16);
}

__device__ __forceinline__ float gelu_f(float x) {
  return 0.5f * x * (1.f + erff(x * 0.70710678118654752f));
}

#define ACT_NONE 0
#define ACT_GELU 1
#define ACT_RELU 2

// ===========================================================================
// Generic batched GEMM: C[M,N] = act(alpha * A * B + bias)
//   A: bf16 row-major, lda = Kpad (mult of 32), M = gridDim.y*64 (exact).
//   B: bf16 N-major, B[n*ldb + k], rows padded to >= gridDim.x*64, k-pad zeroed.
//   C: f32 row-major ldc; store guarded by n < N only.
// Block = 128 threads (4 waves); wave computes 16(M) x 64(N); block tile 64x64.
// Software-pipelined: fragments for step i+1 are in flight while step i's
// 4 WMMAs execute.
// ===========================================================================
__global__ __launch_bounds__(128) void gemm_bf16_wmma(
    const unsigned short* __restrict__ A, const unsigned short* __restrict__ B,
    const float* __restrict__ bias, float* __restrict__ C,
    int N, int KP, int lda, int ldb, int ldc,
    long aBatch, long bBatch, long cBatch, float alpha, int act)
{
  const int lane = threadIdx.x & 31;
  const int wave = threadIdx.x >> 5;
  const int tileN = blockIdx.x * 64;
  const int tileM = blockIdx.y * 64;
  const int ln15  = lane & 15;
  const int kbase = (lane >> 4) << 3;           // 0 or 8

  const unsigned short* pA = A + (long)blockIdx.z * aBatch
                           + (long)(tileM + (wave << 4) + ln15) * lda + kbase;
  const unsigned short* pB = B + (long)blockIdx.z * bBatch
                           + (long)(tileN + ln15) * ldb + kbase;
  const long bStep = (long)ldb << 4;            // 16 rows of B per j-tile

  v8f acc[4];
  #pragma unroll
  for (int j = 0; j < 4; j++) acc[j] = zero8();

  // prologue: fragments for k-step 0
  v16bf va = load_frag(pA);
  v16bf vb0 = load_frag(pB);
  v16bf vb1 = load_frag(pB + bStep);
  v16bf vb2 = load_frag(pB + 2 * bStep);
  v16bf vb3 = load_frag(pB + 3 * bStep);

  for (int kp = 32; kp < KP; kp += 32) {
    // issue next k-step loads before consuming current fragments
    v16bf na  = load_frag(pA + kp);
    v16bf nb0 = load_frag(pB + kp);
    v16bf nb1 = load_frag(pB + bStep + kp);
    v16bf nb2 = load_frag(pB + 2 * bStep + kp);
    v16bf nb3 = load_frag(pB + 3 * bStep + kp);
    __builtin_prefetch(pA + kp + 32, 0, 1);      // global_prefetch_b8 (L2 warmup)
    acc[0] = __builtin_amdgcn_wmma_f32_16x16x32_bf16(false, va, false, vb0, (short)0, acc[0], false, false);
    acc[1] = __builtin_amdgcn_wmma_f32_16x16x32_bf16(false, va, false, vb1, (short)0, acc[1], false, false);
    acc[2] = __builtin_amdgcn_wmma_f32_16x16x32_bf16(false, va, false, vb2, (short)0, acc[2], false, false);
    acc[3] = __builtin_amdgcn_wmma_f32_16x16x32_bf16(false, va, false, vb3, (short)0, acc[3], false, false);
    va = na; vb0 = nb0; vb1 = nb1; vb2 = nb2; vb3 = nb3;
  }
  // epilogue k-step
  acc[0] = __builtin_amdgcn_wmma_f32_16x16x32_bf16(false, va, false, vb0, (short)0, acc[0], false, false);
  acc[1] = __builtin_amdgcn_wmma_f32_16x16x32_bf16(false, va, false, vb1, (short)0, acc[1], false, false);
  acc[2] = __builtin_amdgcn_wmma_f32_16x16x32_bf16(false, va, false, vb2, (short)0, acc[2], false, false);
  acc[3] = __builtin_amdgcn_wmma_f32_16x16x32_bf16(false, va, false, vb3, (short)0, acc[3], false, false);

  const long zC = (long)blockIdx.z * cBatch;
  const int mBase = tileM + (wave << 4) + ((lane >> 4) << 3);
  #pragma unroll
  for (int j = 0; j < 4; j++) {
    int nCol = tileN + (j << 4) + ln15;
    if (nCol >= N) continue;
    float bv = bias ? bias[nCol] : 0.f;
    #pragma unroll
    for (int r = 0; r < 8; r++) {
      float v = acc[j][r] * alpha + bv;
      if (act == ACT_GELU)      v = gelu_f(v);
      else if (act == ACT_RELU) v = fmaxf(v, 0.f);
      C[zC + (long)(mBase + r) * ldc + nCol] = v;
    }
  }
}

// ===========================================================================
// 64x64x64 batched WMMA for encoder attention. blockIdx.x = z = bt*8 + h.
// All operands N-major / k-contiguous => branch-free b128 loads. K fixed at 64
// so all 10 fragments are loaded up front, then 8 WMMAs issue back-to-back.
//   aHead: A[bt*32768 + m*512 + h*64 + k]  else A[z*4096 + m*64 + k]
//   bHead: B[bt*bBatch + n*512 + h*64 + k] else B[z*4096 + n*64 + k]
//   cHead: C[bt*32768 + m*512 + h*64 + n]  else C[z*4096 + m*64 + n]
// ===========================================================================
__global__ __launch_bounds__(128) void attn_gemm64_k(
    const unsigned short* __restrict__ A, int aHead,
    const unsigned short* __restrict__ B, int bHead, long bBatch,
    float* __restrict__ C, int cHead, float alpha)
{
  const int lane = threadIdx.x & 31;
  const int wave = threadIdx.x >> 5;
  const int z  = blockIdx.x;
  const int bt = z >> 3;
  const int h  = z & 7;
  const int ln15  = lane & 15;
  const int kbase = (lane >> 4) << 3;
  const int m = (wave << 4) + ln15;

  const unsigned short* pA = aHead
      ? A + (long)bt * 32768 + (long)m * 512 + h * 64 + kbase
      : A + ((long)z << 12) + m * 64 + kbase;
  const unsigned short* pB = bHead
      ? B + (long)bt * bBatch + (long)ln15 * 512 + h * 64 + kbase
      : B + ((long)z << 12) + ln15 * 64 + kbase;
  const long bStep = bHead ? (512L << 4) : (64L << 4);

  // load both k-steps up front (10 fragments = 20 b128 loads)
  v16bf a0  = load_frag(pA);
  v16bf a1  = load_frag(pA + 32);
  v16bf b00 = load_frag(pB);
  v16bf b01 = load_frag(pB + bStep);
  v16bf b02 = load_frag(pB + 2 * bStep);
  v16bf b03 = load_frag(pB + 3 * bStep);
  v16bf b10 = load_frag(pB + 32);
  v16bf b11 = load_frag(pB + bStep + 32);
  v16bf b12 = load_frag(pB + 2 * bStep + 32);
  v16bf b13 = load_frag(pB + 3 * bStep + 32);

  v8f acc[4];
  #pragma unroll
  for (int j = 0; j < 4; j++) acc[j] = zero8();

  acc[0] = __builtin_amdgcn_wmma_f32_16x16x32_bf16(false, a0, false, b00, (short)0, acc[0], false, false);
  acc[1] = __builtin_amdgcn_wmma_f32_16x16x32_bf16(false, a0, false, b01, (short)0, acc[1], false, false);
  acc[2] = __builtin_amdgcn_wmma_f32_16x16x32_bf16(false, a0, false, b02, (short)0, acc[2], false, false);
  acc[3] = __builtin_amdgcn_wmma_f32_16x16x32_bf16(false, a0, false, b03, (short)0, acc[3], false, false);
  acc[0] = __builtin_amdgcn_wmma_f32_16x16x32_bf16(false, a1, false, b10, (short)0, acc[0], false, false);
  acc[1] = __builtin_amdgcn_wmma_f32_16x16x32_bf16(false, a1, false, b11, (short)0, acc[1], false, false);
  acc[2] = __builtin_amdgcn_wmma_f32_16x16x32_bf16(false, a1, false, b12, (short)0, acc[2], false, false);
  acc[3] = __builtin_amdgcn_wmma_f32_16x16x32_bf16(false, a1, false, b13, (short)0, acc[3], false, false);

  const int mB = (wave << 4) + ((lane >> 4) << 3);
  #pragma unroll
  for (int j = 0; j < 4; j++) {
    int n = (j << 4) + ln15;
    #pragma unroll
    for (int r = 0; r < 8; r++) {
      int mm = mB + r;
      float v = acc[j][r] * alpha;
      if (cHead) C[(long)bt * 32768 + (long)mm * 512 + h * 64 + n] = v;
      else       C[((long)z << 12) + mm * 64 + n] = v;
    }
  }
}

// ===========================================================================
// Conversion / staging kernels
// ===========================================================================
__global__ void f32_to_bf16_k(const float* __restrict__ in,
                              unsigned short* __restrict__ out, long n)
{
  long i = (long)blockIdx.x * blockDim.x + threadIdx.x;
  long stride = (long)gridDim.x * blockDim.x;
  for (; i < n; i += stride) out[i] = f2bf(in[i]);
}

// A staging with K padding: out[m*Kpad + k] = (k<K) ? bf16(in[m*K + k]) : 0
__global__ void convA_pad_k(const float* __restrict__ in,
                            unsigned short* __restrict__ out,
                            long n, int K, int Kpad)
{
  long i = (long)blockIdx.x * blockDim.x + threadIdx.x;
  long stride = (long)gridDim.x * blockDim.x;
  for (; i < n; i += stride) {
    long m = i / Kpad; int k = (int)(i % Kpad);
    out[i] = (k < K) ? f2bf(in[m * (long)K + k]) : (unsigned short)0;
  }
}

// Weight transpose to padded N-major: out[n*Kpad + k] = bf16(in[k*N + n]), pad 0
__global__ void convW_T_k(const float* __restrict__ in,
                          unsigned short* __restrict__ out,
                          long n_total, int K, int N, int Kpad)
{
  long i = (long)blockIdx.x * blockDim.x + threadIdx.x;
  long stride = (long)gridDim.x * blockDim.x;
  for (; i < n_total; i += stride) {
    long n = i / Kpad; int k = (int)(i % Kpad);
    out[i] = (k < K && n < N) ? f2bf(in[(long)k * N + n]) : (unsigned short)0;
  }
}

// Per-head transpose of V: out[z*4096 + d*64 + kidx] = bf16(V[bt,kidx,h,d])
__global__ void vtrans_k(const float* __restrict__ v, unsigned short* __restrict__ out)
{
  long i = (long)blockIdx.x * blockDim.x + threadIdx.x;
  long stride = (long)gridDim.x * blockDim.x;
  for (; i < 2048L * 4096; i += stride) {
    long z = i >> 12; int rem = (int)(i & 4095);
    int d = rem >> 6, kk = rem & 63;
    int bt = (int)(z >> 3), h = (int)(z & 7);
    out[i] = f2bf(v[(long)bt * 32768 + (long)kk * 512 + h * 64 + d]);
  }
}

// ===========================================================================
// Elementwise / glue kernels
// ===========================================================================
__global__ void add_row_k(const float* __restrict__ x, const float* __restrict__ vec,
                          float* __restrict__ out, long n, int cols)
{
  long i = (long)blockIdx.x * blockDim.x + threadIdx.x;
  long stride = (long)gridDim.x * blockDim.x;
  for (; i < n; i += stride) out[i] = x[i] + vec[i % cols];
}

__global__ void add_post_k(float* __restrict__ x, const float* __restrict__ pos,
                           long n, int cols, int tmod)
{
  long i = (long)blockIdx.x * blockDim.x + threadIdx.x;
  long stride = (long)gridDim.x * blockDim.x;
  for (; i < n; i += stride) {
    long r = i / cols; int c = (int)(i % cols);
    x[i] += pos[(long)(r % tmod) * cols + c];
  }
}

__global__ void magating_k(float* __restrict__ h, const float* __restrict__ a,
                           const float* __restrict__ b, long n)
{
  long i = (long)blockIdx.x * blockDim.x + threadIdx.x;
  long stride = (long)gridDim.x * blockDim.x;
  for (; i < n; i += stride) {
    int p = (int)(i % 32768);
    h[i] = h[i] * __expf(a[p]) + b[p];
  }
}

__global__ void sinusoid_k(float* __restrict__ pe)
{
  long i = (long)blockIdx.x * blockDim.x + threadIdx.x;
  if (i >= 32768) return;
  int s = (int)(i >> 9), d = (int)(i & 511);
  int i2 = d & ~1;
  float freq = powf(10000.f, -(float)i2 / 512.f);
  float ang = (float)s * freq;
  pe[i] = (d & 1) ? cosf(ang) : sinf(ang);
}

__global__ void board_gather_k(const float* __restrict__ h, float* __restrict__ board)
{
  long i = (long)blockIdx.x * blockDim.x + threadIdx.x;
  if (i >= 65536) return;
  int r = (int)(i >> 8), c = (int)(i & 255);
  int s = c >> 2, ch = c & 3;
  board[i] = h[((long)r * 64 + s) * 512 + ch];
}

// Transformer-XL relative shift + combine + scale (in-place into `content`)
__global__ void relshift_k(float* __restrict__ content, const float* __restrict__ ps,
                           float alpha)
{
  long i = (long)blockIdx.x * blockDim.x + threadIdx.x;
  if (i >= 8388608L) return;
  long z = i >> 12; int qc = (int)(i & 4095);
  int q = qc >> 6, c = qc & 63;
  int flat = (q + 1) * 64 + c;
  int ii = flat / 65, jj = flat % 65;
  float sv = (jj == 0) ? 0.f : ps[(z << 12) + ii * 64 + (jj - 1)];
  content[i] = (content[i] + sv) * alpha;
}

__global__ void softmax64_k(float* __restrict__ x, long rows)
{
  long row = (long)blockIdx.x * (blockDim.x >> 5) + (threadIdx.x >> 5);
  if (row >= rows) return;
  int lane = threadIdx.x & 31;
  float* p = x + (row << 6);
  float a = p[lane], b = p[lane + 32];
  float mx = fmaxf(a, b);
  #pragma unroll
  for (int o = 16; o > 0; o >>= 1) mx = fmaxf(mx, __shfl_xor(mx, o, 32));
  a = __expf(a - mx); b = __expf(b - mx);
  float s = a + b;
  #pragma unroll
  for (int o = 16; o > 0; o >>= 1) s += __shfl_xor(s, o, 32);
  float inv = 1.f / s;
  p[lane] = a * inv; p[lane + 32] = b * inv;
}

__global__ void layernorm_k(const float* __restrict__ x, const float* __restrict__ res,
                            const float* __restrict__ g, const float* __restrict__ b,
                            float* __restrict__ out, int cols)
{
  __shared__ float buf[2048];
  __shared__ float red[8];
  long row = blockIdx.x;
  const float* px = x + row * (long)cols;
  const float* pr = res ? res + row * (long)cols : (const float*)0;
  float s = 0.f;
  for (int c = threadIdx.x; c < cols; c += blockDim.x) {
    float v = px[c] + (pr ? pr[c] : 0.f);
    buf[c] = v; s += v;
  }
  #pragma unroll
  for (int o = 16; o > 0; o >>= 1) s += __shfl_xor(s, o, 32);
  if ((threadIdx.x & 31) == 0) red[threadIdx.x >> 5] = s;
  __syncthreads();
  float mean = 0.f;
  int nw = (int)(blockDim.x >> 5);
  for (int i = 0; i < nw; i++) mean += red[i];
  mean /= (float)cols;
  __syncthreads();
  float vs = 0.f;
  for (int c = threadIdx.x; c < cols; c += blockDim.x) {
    float d = buf[c] - mean; vs += d * d;
  }
  #pragma unroll
  for (int o = 16; o > 0; o >>= 1) vs += __shfl_xor(vs, o, 32);
  if ((threadIdx.x & 31) == 0) red[threadIdx.x >> 5] = vs;
  __syncthreads();
  float var = 0.f;
  for (int i = 0; i < nw; i++) var += red[i];
  var /= (float)cols;
  float inv = rsqrtf(var + 1e-5f);
  float* po = out + row * (long)cols;
  for (int c = threadIdx.x; c < cols; c += blockDim.x)
    po[c] = (buf[c] - mean) * inv * g[c] + b[c];
}

// ---- tiny decoder attention over history axis (t=8) ----
__global__ void dec_scores_k(const float* __restrict__ q, const float* __restrict__ k,
                             float* __restrict__ sc)
{
  int lane = threadIdx.x & 31;
  int w = blockIdx.x * (blockDim.x >> 5) + (threadIdx.x >> 5);
  if (w >= 2048) return;
  int b = w >> 6, ij = w & 63, i = ij >> 3, j = ij & 7;
  const float* pq = q + (long)(b * 8 + i) * 1929;
  const float* pk = k + (long)(b * 8 + j) * 1929;
  float s = 0.f;
  for (int c = lane; c < 1929; c += 32) s += pq[c] * pk[c];
  #pragma unroll
  for (int o = 16; o > 0; o >>= 1) s += __shfl_xor(s, o, 32);
  if (lane == 0) sc[w] = (j > i) ? -1e9f : s * rsqrtf(1929.f);
}

__global__ void dec_softmax8_k(float* __restrict__ sc)
{
  int tid = blockIdx.x * blockDim.x + threadIdx.x;
  if (tid >= 256) return;
  int b = tid >> 3, i = tid & 7;
  float* p = sc + b * 64 + i * 8;
  float mx = p[0];
  #pragma unroll
  for (int j = 1; j < 8; j++) mx = fmaxf(mx, p[j]);
  float s = 0.f, e[8];
  #pragma unroll
  for (int j = 0; j < 8; j++) { e[j] = __expf(p[j] - mx); s += e[j]; }
  float inv = 1.f / s;
  #pragma unroll
  for (int j = 0; j < 8; j++) p[j] = e[j] * inv;
}

__global__ void dec_av_k(const float* __restrict__ attn, const float* __restrict__ v,
                         float* __restrict__ ao)
{
  long idx = (long)blockIdx.x * blockDim.x + threadIdx.x;
  if (idx >= 256L * 1929) return;
  int r = (int)(idx / 1929), d = (int)(idx % 1929);
  int b = r >> 3, i = r & 7;
  const float* pa = attn + b * 64 + i * 8;
  float s = 0.f;
  #pragma unroll
  for (int j = 0; j < 8; j++) s += pa[j] * v[(long)(b * 8 + j) * 1929 + d];
  ao[idx] = s;
}

// ===========================================================================
// Host orchestration
// ===========================================================================
static inline int gs_blocks(long n) {
  long b = (n + 255) / 256;
  return (int)(b > 16384 ? 16384 : b);
}

extern "C" void kernel_launch(void* const* d_in, const int* in_sizes, int n_in,
                              void* d_out, int out_size, void* d_ws, size_t ws_size,
                              hipStream_t stream)
{
  (void)in_sizes; (void)n_in; (void)out_size; (void)ws_size;

  // ---- input index map (jax pytree flatten: dict keys sorted) ----
  auto DEC = [&](int l, int f) { return (const float*)d_in[l * 16 + f]; };
  auto LYR = [&](int l, int f) { return (const float*)d_in[66 + l * 19 + f]; };
  enum { dF1B=0,dF1W,dF2B,dF2W,dKB,dKW,dLN1B,dLN1G,dLN2B,dLN2G,dOB,dOW,dQB,dQW,dVB,dVW };
  enum { eB1=0,eB2,eBK,eBO,eBQ,eBV,eLN1B,eLN1G,eLN2B,eLN2G,eU,eVB,eW1,eW2,eWK,eWO,eWPOS,eWQ,eWV };
  const int KPB=64, KPW=65, LIN1B=256, LIN1W=257, LN1B=258, LN1G=259,
            MA_A=260, MA_B=261, PHW=262, POS_BOARD=263, POS_POLICY=264,
            PTB=265, PTW=266, QPB=267, QPW=268, XIN=269;

  const float RSQ_D = 0.044194173824159216f;   // 1/sqrt(512)

  // ---- workspace carve (deterministic bump allocator) ----
  char* wp = (char*)d_ws;
  auto carve = [&](size_t bytes) -> char* {
    char* p = wp; wp += (bytes + 255) & ~(size_t)255; return p;
  };
  float* H0    = (float*)carve(16384UL * 512 * 4);
  float* TMP1  = (float*)carve(16384UL * 736 * 4);
  float* Qb    = (float*)carve(16384UL * 512 * 4);
  float* Kb    = (float*)carve(16384UL * 512 * 4);
  float* Vb    = (float*)carve(16384UL * 512 * 4);
  float* QB    = (float*)carve(16384UL * 512 * 4);   // Qu/Qv staging, ctx, ff out
  float* SC1   = (float*)carve(2048UL * 4096 * 4);
  float* SC2   = (float*)carve(2048UL * 4096 * 4);
  unsigned short* ABF = (unsigned short*)carve(16384UL * 736 * 2);
  unsigned short* BBF = (unsigned short*)carve(16384UL * 512 * 2);  // 16.8 MB
  float* POSF  = (float*)carve(64UL * 512 * 4);
  float* PPROJ = (float*)carve(64UL * 512 * 4);
  unsigned short* POSBF = (unsigned short*)carve(64UL * 512 * 2);
  unsigned short* PPBF  = (unsigned short*)carve(64UL * 512 * 2);
  float* MMb   = (float*)carve(256UL * 4096 * 4);
  float* P0    = (float*)carve(256UL * 1929 * 4);
  float* P1    = (float*)carve(256UL * 1929 * 4);
  float* BOARD = (float*)carve(256UL * 256 * 4);
  float* QD    = (float*)carve(256UL * 1929 * 4);
  float* KD    = (float*)carve(256UL * 1929 * 4);
  float* VD    = (float*)carve(256UL * 1929 * 4);
  float* AOD   = (float*)carve(256UL * 1929 * 4);
  float* TD    = (float*)carve(256UL * 1929 * 4);
  float* FFD   = (float*)carve(256UL * 3858 * 4);
  float* SCD   = (float*)carve(2048UL * 4);

  // ---- launch helpers ----
  auto conv = [&](const float* src, unsigned short* dst, long n) {
    f32_to_bf16_k<<<gs_blocks(n), 256, 0, stream>>>(src, dst, n);
  };
  auto convA = [&](const float* src, unsigned short* dst, long M, int K, int Kpad) {
    long n = M * Kpad;
    convA_pad_k<<<gs_blocks(n), 256, 0, stream>>>(src, dst, n, K, Kpad);
  };
  auto convW = [&](const float* src, unsigned short* dst, int K, int N,
                   int Kpad, int Npad) {
    long n = (long)Npad * Kpad;
    convW_T_k<<<gs_blocks(n), 256, 0, stream>>>(src, dst, n, K, N, Kpad);
  };
  auto gemm = [&](const unsigned short* A, const unsigned short* B, const float* bias,
                  float* C, int M, int N, int KP, int lda, int ldb, int ldc,
                  long aB, long bB, long cB, int batches, float alpha, int act) {
    dim3 g((N + 63) / 64, M / 64, batches);
    gemm_bf16_wmma<<<g, 128, 0, stream>>>(A, B, bias, C, N, KP, lda, ldb, ldc,
                                          aB, bB, cB, alpha, act);
  };
  auto ln = [&](const float* x, const float* res, const float* g, const float* b,
                float* out, long rows, int cols) {
    layernorm_k<<<(unsigned)rows, 256, 0, stream>>>(x, res, g, b, out, cols);
  };

  // =========================== embedding ===========================
  sinusoid_k<<<128, 256, 0, stream>>>(POSF);
  conv(POSF, POSBF, 64L * 512);

  convA((const float*)d_in[XIN], ABF, 16384, 19, 32);
  convW((const float*)d_in[LIN1W], BBF, 19, 512, 32, 512);
  gemm(ABF, BBF, (const float*)d_in[LIN1B], H0,
       16384, 512, 32, 32, 32, 512, 0, 0, 0, 1, 1.f, ACT_GELU);
  ln(H0, nullptr, (const float*)d_in[LN1G], (const float*)d_in[LN1B], H0, 16384, 512);
  magating_k<<<gs_blocks(16384L * 512), 256, 0, stream>>>(
      H0, (const float*)d_in[MA_A], (const float*)d_in[MA_B], 16384L * 512);

  // =========================== encoder x10 ===========================
  for (int l = 0; l < 10; l++) {
    conv(H0, ABF, 16384L * 512);
    convW(LYR(l, eWQ), BBF, 512, 512, 512, 512);
    gemm(ABF, BBF, LYR(l, eBQ), Qb, 16384, 512, 512, 512, 512, 512, 0,0,0,1, 1.f, ACT_NONE);
    convW(LYR(l, eWK), BBF, 512, 512, 512, 512);
    gemm(ABF, BBF, LYR(l, eBK), Kb, 16384, 512, 512, 512, 512, 512, 0,0,0,1, 1.f, ACT_NONE);
    convW(LYR(l, eWV), BBF, 512, 512, 512, 512);
    gemm(ABF, BBF, LYR(l, eBV), Vb, 16384, 512, 512, 512, 512, 512, 0,0,0,1, 1.f, ACT_NONE);

    // shared positional projection pp = pos @ wpos (no bias)
    convW(LYR(l, eWPOS), BBF, 512, 512, 512, 512);
    gemm(POSBF, BBF, nullptr, PPROJ, 64, 512, 512, 512, 512, 512, 0,0,0,1, 1.f, ACT_NONE);
    conv(PPROJ, PPBF, 64L * 512);

    // content scores: (q + u) . k    (K rows are already N-major, k-contiguous)
    add_row_k<<<gs_blocks(16384L * 512), 256, 0, stream>>>(Qb, LYR(l, eU), QB,
                                                           16384L * 512, 512);
    conv(QB, ABF, 16384L * 512);
    conv(Kb, BBF, 16384L * 512);
    attn_gemm64_k<<<2048, 128, 0, stream>>>(ABF, 1, BBF, 1, 32768L, SC1, 0, 1.f);

    // positional scores: (q + vbias) . pp
    add_row_k<<<gs_blocks(16384L * 512), 256, 0, stream>>>(Qb, LYR(l, eVB), QB,
                                                           16384L * 512, 512);
    conv(QB, ABF, 16384L * 512);
    attn_gemm64_k<<<2048, 128, 0, stream>>>(ABF, 1, PPBF, 1, 0L, SC2, 0, 1.f);

    // rel-shift + combine + scale, softmax over keys
    relshift_k<<<32768, 256, 0, stream>>>(SC1, SC2, RSQ_D);
    softmax64_k<<<16384, 256, 0, stream>>>(SC1, 131072L);

    // ctx = attn @ V  (V transposed per-head so B is N-major)
    conv(SC1, ABF, 2048L * 4096);
    vtrans_k<<<gs_blocks(2048L * 4096), 256, 0, stream>>>(Vb, BBF);
    attn_gemm64_k<<<2048, 128, 0, stream>>>(ABF, 0, BBF, 0, 0L, QB, 1, 1.f);

    // out proj + residual LN
    conv(QB, ABF, 16384L * 512);
    convW(LYR(l, eWO), BBF, 512, 512, 512, 512);
    gemm(ABF, BBF, LYR(l, eBO), TMP1, 16384, 512, 512, 512, 512, 512, 0,0,0,1, 1.f, ACT_NONE);
    ln(TMP1, H0, LYR(l, eLN1G), LYR(l, eLN1B), H0, 16384, 512);

    // feed-forward: y = gelu(ff2(ff1(x))) ; ln(y + x)
    conv(H0, ABF, 16384L * 512);
    convW(LYR(l, eW1), BBF, 512, 736, 512, 768);
    gemm(ABF, BBF, LYR(l, eB1), TMP1, 16384, 736, 512, 512, 512, 736, 0,0,0,1, 1.f, ACT_NONE);
    conv(TMP1, ABF, 16384L * 736);
    convW(LYR(l, eW2), BBF, 736, 512, 736, 512);
    gemm(ABF, BBF, LYR(l, eB2), QB, 16384, 512, 736, 736, 736, 512, 0,0,0,1, 1.f, ACT_GELU);
    ln(QB, H0, LYR(l, eLN2G), LYR(l, eLN2B), H0, 16384, 512);
  }

  // =========================== heads ===========================
  board_gather_k<<<256, 256, 0, stream>>>(H0, BOARD);

  conv(H0, ABF, 16384L * 512);
  convW((const float*)d_in[PTW], BBF, 512, 512, 512, 512);
  gemm(ABF, BBF, (const float*)d_in[PTB], Qb, 16384, 512, 512, 512, 512, 512,
       0,0,0,1, 1.f, ACT_GELU);                                  // pt in Qb
  conv(Qb, ABF, 16384L * 512);
  convW((const float*)d_in[QPW], BBF, 512, 512, 512, 512);
  gemm(ABF, BBF, (const float*)d_in[QPB], Kb, 16384, 512, 512, 512, 512, 512,
       0,0,0,1, 1.f, ACT_NONE);                                  // qp in Kb
  convW((const float*)d_in[KPW], BBF, 512, 512, 512, 512);
  gemm(ABF, BBF, (const float*)d_in[KPB], Vb, 16384, 512, 512, 512, 512, 512,
       0,0,0,1, 1.f, ACT_NONE);                                  // kp in Vb

  // mm[bt,q,k] = qp . kp / sqrt(512)  (kp rows already N-major; 256 batches)
  conv(Kb, ABF, 16384L * 512);
  conv(Vb, BBF, 16384L * 512);
  gemm(ABF, BBF, nullptr, MMb, 64, 64, 512, 512, 512, 64,
       32768L, 32768L, 4096L, 256, RSQ_D, ACT_NONE);

  // pol = mm.reshape(256, 4096) @ phw(4096x1929)
  conv(MMb, ABF, 256L * 4096);
  convW((const float*)d_in[PHW], BBF, 4096, 1929, 4096, 1984);
  gemm(ABF, BBF, nullptr, P0, 256, 1929, 4096, 4096, 4096, 1929, 0,0,0,1, 1.f, ACT_NONE);

  add_post_k<<<gs_blocks(256L * 1929), 256, 0, stream>>>(
      P0, (const float*)d_in[POS_POLICY], 256L * 1929, 1929, 8);
  add_post_k<<<gs_blocks(256L * 256), 256, 0, stream>>>(
      BOARD, (const float*)d_in[POS_BOARD], 256L * 256, 256, 8);

  // =========================== decoder x4 ===========================
  float* pe = P0;
  for (int l = 0; l < 4; l++) {
    convA(pe, ABF, 256, 1929, 1952);
    convW(DEC(l, dQW), BBF, 1929, 1929, 1952, 1984);
    gemm(ABF, BBF, DEC(l, dQB), QD, 256, 1929, 1952, 1952, 1952, 1929, 0,0,0,1, 1.f, ACT_NONE);
    convW(DEC(l, dVW), BBF, 1929, 1929, 1952, 1984);
    gemm(ABF, BBF, DEC(l, dVB), VD, 256, 1929, 1952, 1952, 1952, 1929, 0,0,0,1, 1.f, ACT_NONE);
    conv(BOARD, ABF, 256L * 256);
    convW(DEC(l, dKW), BBF, 256, 1929, 256, 1984);
    gemm(ABF, BBF, DEC(l, dKB), KD, 256, 1929, 256, 256, 256, 1929, 0,0,0,1, 1.f, ACT_NONE);

    dec_scores_k<<<512, 128, 0, stream>>>(QD, KD, SCD);
    dec_softmax8_k<<<1, 256, 0, stream>>>(SCD);
    dec_av_k<<<gs_blocks(256L * 1929), 256, 0, stream>>>(SCD, VD, AOD);

    convA(AOD, ABF, 256, 1929, 1952);
    convW(DEC(l, dOW), BBF, 1929, 1929, 1952, 1984);
    gemm(ABF, BBF, DEC(l, dOB), TD, 256, 1929, 1952, 1952, 1952, 1929, 0,0,0,1, 1.f, ACT_NONE);
    ln(TD, pe, DEC(l, dLN1G), DEC(l, dLN1B), P1, 256, 1929);

    convA(P1, ABF, 256, 1929, 1952);
    convW(DEC(l, dF1W), BBF, 1929, 3858, 1952, 3904);
    gemm(ABF, BBF, DEC(l, dF1B), FFD, 256, 3858, 1952, 1952, 1952, 3858, 0,0,0,1, 1.f, ACT_RELU);
    convA(FFD, ABF, 256, 3858, 3872);
    convW(DEC(l, dF2W), BBF, 3858, 1929, 3872, 1984);
    gemm(ABF, BBF, DEC(l, dF2B), TD, 256, 1929, 3872, 3872, 3872, 1929, 0,0,0,1, 1.f, ACT_NONE);

    float* outp = (l == 3) ? (float*)d_out : P0;
    ln(TD, P1, DEC(l, dLN2G), DEC(l, dLN2B), outp, 256, 1929);
    pe = P0;
  }
}
